// Attention_14714557956672
// MI455X (gfx1250) — matrix operand order repaired and checked by reading, not compiled, not run on hardware
//
#include <hip/hip_runtime.h>

// ---------------------------------------------------------------------------
// Types for CDNA5 WMMA (wave32): 16x16x32 bf16 -> f32
// ---------------------------------------------------------------------------
typedef __attribute__((ext_vector_type(16))) __bf16 v16bf;
typedef __attribute__((ext_vector_type(8)))  float  v8f;
typedef __attribute__((ext_vector_type(4)))  int    v4i;

union ABFrag { unsigned int u[8]; v16bf v; };

__device__ __forceinline__ unsigned short f32_to_bf16(float f) {
  unsigned int u = __builtin_bit_cast(unsigned int, f);
  u = (u + 0x7FFFu + ((u >> 16) & 1u)) >> 16;   // round-to-nearest-even
  return (unsigned short)u;
}

// ---------------------------------------------------------------------------
// Async global->LDS copy (CDNA5 GLOBAL_LOAD_ASYNC_TO_LDS_B128, ASYNCcnt).
// Builtin signature (from hipcc diagnostic): (v4i AS1*, v4i AS3*, imm, imm).
// Guarded: falls back to a synchronous copy if the builtin is unavailable.
// ---------------------------------------------------------------------------
#define AS_G __attribute__((address_space(1)))
#define AS_L __attribute__((address_space(3)))

#if defined(__AMDGCN__) &&                                                  \
    __has_builtin(__builtin_amdgcn_global_load_async_to_lds_b128) &&        \
    __has_builtin(__builtin_amdgcn_s_wait_asynccnt)
#define USE_ASYNC 1
#else
#define USE_ASYNC 0
#endif

__device__ __forceinline__ void cp_b128(void* lds, const void* gptr) {
#if USE_ASYNC
  __builtin_amdgcn_global_load_async_to_lds_b128(
      (AS_G v4i*)gptr, (AS_L v4i*)lds, 0, 0);
#else
  *(uint4*)lds = *(const uint4*)gptr;
#endif
}
__device__ __forceinline__ void cp_wait() {
#if USE_ASYNC
  __builtin_amdgcn_s_wait_asynccnt(0);
#endif
}

// ---------------------------------------------------------------------------
// Problem constants
// ---------------------------------------------------------------------------
#define BATCH   2
#define SEQ     2048
#define CDIM    1024
#define NHEADS  16
#define HDIM    64
#define MTOT    (BATCH * SEQ)          // 4096 rows
#define NEG_INF (-3.0e38f)

// ---------------------------------------------------------------------------
// K0a: fp32 -> bf16 pair converter (row-major copy, for x)
// ---------------------------------------------------------------------------
__global__ void f32_to_bf16_kernel(const float* __restrict__ src,
                                   unsigned int* __restrict__ dst, int npairs) {
  int i = blockIdx.x * blockDim.x + threadIdx.x;
  if (i < npairs) {
    unsigned int lo = f32_to_bf16(src[2 * i]);
    unsigned int hi = f32_to_bf16(src[2 * i + 1]);
    dst[i] = lo | (hi << 16);
  }
}

// ---------------------------------------------------------------------------
// K0b: fp32 (k, col) -> bf16 (col, kpair) transposed converter for weights.
// dst row stride = rows/2 uints.  One-time cost; makes all GEMM B-tile
// staging a straight (async) b128 copy.
// ---------------------------------------------------------------------------
__global__ void convert_transpose_kernel(const float* __restrict__ src,
                                         unsigned int* __restrict__ dst,
                                         int rows, int cols) {
  int idx = blockIdx.x * blockDim.x + threadIdx.x;   // over cols*rows/2
  int total = cols * (rows >> 1);
  if (idx < total) {
    int kp  = idx / cols;
    int col = idx - kp * cols;
    unsigned int lo = f32_to_bf16(src[(size_t)(2 * kp) * cols + col]);
    unsigned int hi = f32_to_bf16(src[(size_t)(2 * kp + 1) * cols + col]);
    dst[(size_t)col * (rows >> 1) + kp] = lo | (hi << 16);
  }
}

// ---------------------------------------------------------------------------
// K1: QKV GEMM (4096x1024 @ 1024x3072) + bias + per-head RMSNorm + RoPE.
// Block = 128 thr (4 waves), tile = 64x64 (one head column-block), K-step 32.
// Double-buffered async LDS staging. V is written TRANSPOSED (b,h,d,n).
// ---------------------------------------------------------------------------
__global__ __launch_bounds__(128) void qkv_gemm_kernel(
    const unsigned int* __restrict__ xa,    // bf16 x, (4096 x 512) uints
    const unsigned int* __restrict__ wT,    // bf16 W_qkv^T, (3072 x 512) uints
    const float* __restrict__ bias,         // 3072
    const float* __restrict__ rope_cos,     // (2048 x 32)
    const float* __restrict__ rope_sin,     // (2048 x 32)
    const float* __restrict__ q_norm_w,     // 64
    const float* __restrict__ k_norm_w,     // 64
    unsigned short* __restrict__ q_out,     // (32, 2048, 64) bf16
    unsigned short* __restrict__ k_out,     // (32, 2048, 64) bf16
    unsigned short* __restrict__ v_outT) {  // (32, 64, 2048) bf16 (transposed)
  __shared__ __align__(16) unsigned int lds_a[2][64 * 16];  // (row, kpair)
  __shared__ __align__(16) unsigned int lds_b[2][64 * 16];  // (col, kpair)

  const int tid  = threadIdx.x;
  const int wave = tid >> 5;
  const int lane = tid & 31;
  const int ln   = lane & 15;
  const int hh   = lane >> 4;
  const int rowBase = blockIdx.y * 64;
  const int colBase = blockIdx.x * 64;

  v8f acc[4];
  #pragma unroll
  for (int t = 0; t < 4; t++)
    #pragma unroll
    for (int j = 0; j < 8; j++) acc[t][j] = 0.0f;

  // stage tile kb into buffer buf (straight b128 copies, async if available)
  auto stage = [&](int kb, int buf) {
    const int kp0 = kb * 16;                         // k0/2
    #pragma unroll
    for (int i = 0; i < 2; i++) {
      int u = i * 128 + tid;                         // 0..255
      int r = u >> 2, q = (u & 3) * 4;
      cp_b128(&lds_a[buf][r * 16 + q],
              &xa[(size_t)(rowBase + r) * 512 + kp0 + q]);
      cp_b128(&lds_b[buf][r * 16 + q],
              &wT[(size_t)(colBase + r) * 512 + kp0 + q]);
    }
  };

  stage(0, 0);
  cp_wait();
  __syncthreads();

  for (int kb = 0; kb < 32; kb++) {
    const int cur = kb & 1;
    if (kb + 1 < 32) stage(kb + 1, cur ^ 1);

    ABFrag af;
    #pragma unroll
    for (int j = 0; j < 8; j++)   // A layout: K = (j&3)*2 + h*8 + (j>>2)*16
      af.u[j] = lds_a[cur][(wave * 16 + ln) * 16 + (j & 3) + hh * 4 + (j >> 2) * 8];
    #pragma unroll
    for (int t = 0; t < 4; t++) {
      ABFrag bf;
      #pragma unroll
      for (int j = 0; j < 8; j++) // B layout: K = (j&3)*2 + h*8 + (j>>2)*16
        bf.u[j] = lds_b[cur][(t * 16 + ln) * 16 + (j & 3) + hh * 4 + (j >> 2) * 8];
      acc[t] = __builtin_amdgcn_wmma_f32_16x16x32_bf16(
          false, af.v, false, bf.v, (short)0, acc[t], false, false);
    }
    cp_wait();
    __syncthreads();
  }

  // ---- epilogue: bias, then split q/k/v, RMSNorm + RoPE for q/k ----
  const int s    = colBase >> 10;          // 0=q 1=k 2=v
  const int head = (colBase & 1023) >> 6;
  #pragma unroll
  for (int t = 0; t < 4; t++) {
    float b = bias[colBase + t * 16 + ln];
    #pragma unroll
    for (int j = 0; j < 8; j++) acc[t][j] += b;
  }

  if (s == 2) {
    // write V transposed: (bh, d, n)
    #pragma unroll
    for (int j = 0; j < 8; j++) {
      int m = rowBase + wave * 16 + j + 8 * hh;
      int b = m >> 11, n = m & 2047;
      size_t o = ((size_t)(b * NHEADS + head)) * HDIM * SEQ + n;
      #pragma unroll
      for (int t = 0; t < 4; t++)
        v_outT[o + (size_t)(t * 16 + ln) * SEQ] = f32_to_bf16(acc[t][j]);
    }
  } else {
    const float* nw = (s == 0) ? q_norm_w : k_norm_w;
    unsigned short* qk = (s == 0) ? q_out : k_out;
    float w0 = nw[ln], w1 = nw[16 + ln], w2 = nw[32 + ln], w3 = nw[48 + ln];
    #pragma unroll
    for (int j = 0; j < 8; j++) {
      float ss = acc[0][j] * acc[0][j] + acc[1][j] * acc[1][j] +
                 acc[2][j] * acc[2][j] + acc[3][j] * acc[3][j];
      ss += __shfl_xor(ss, 1, 32);
      ss += __shfl_xor(ss, 2, 32);
      ss += __shfl_xor(ss, 4, 32);
      ss += __shfl_xor(ss, 8, 32);
      float rn = rsqrtf(ss * (1.0f / 64.0f) + 1e-6f);
      int m = rowBase + wave * 16 + j + 8 * hh;
      int b = m >> 11, n = m & 2047;
      float c0 = rope_cos[n * 32 + ln], c1 = rope_cos[n * 32 + 16 + ln];
      float s0 = rope_sin[n * 32 + ln], s1 = rope_sin[n * 32 + 16 + ln];
      float x0 = acc[0][j] * rn * w0, x1 = acc[1][j] * rn * w1;
      float x2 = acc[2][j] * rn * w2, x3 = acc[3][j] * rn * w3;
      float o0 = x0 * c0 - x2 * s0;      // d<32 : x*cos - x[d+32]*sin
      float o1 = x1 * c1 - x3 * s1;
      float o2 = x2 * c0 + x0 * s0;      // d>=32: x*cos + x[d-32]*sin
      float o3 = x3 * c1 + x1 * s1;
      size_t o = ((size_t)((b * NHEADS + head) * SEQ + n)) * HDIM;
      qk[o + ln]      = f32_to_bf16(o0);
      qk[o + 16 + ln] = f32_to_bf16(o1);
      qk[o + 32 + ln] = f32_to_bf16(o2);
      qk[o + 48 + ln] = f32_to_bf16(o3);
    }
  }
}

// ---------------------------------------------------------------------------
// K2: flash attention per (b,h). Block = 64 q-rows (4 waves x 16), streams
// 64-key tiles with double-buffered async staging (K straight, V already
// transposed by the producer); online softmax; P re-layout via LDS.
// ---------------------------------------------------------------------------
__global__ __launch_bounds__(128) void flash_attn_kernel(
    const unsigned short* __restrict__ qg,   // (32, 2048, 64)
    const unsigned short* __restrict__ kg,   // (32, 2048, 64)
    const unsigned short* __restrict__ vgT,  // (32, 64, 2048)
    unsigned short* __restrict__ outg) {     // (B, N, H*64) bf16
  __shared__ __align__(16) unsigned int lds_k[2][2048];  // (key, dpair)
  __shared__ __align__(16) unsigned int lds_v[2][2048];  // (d, keypair)
  __shared__ __align__(16) unsigned int lds_p[2048];     // (qrow, keypair)

  const int tid  = threadIdx.x;
  const int wave = tid >> 5;
  const int lane = tid & 31;
  const int ln   = lane & 15;
  const int hh   = lane >> 4;
  const int bh    = blockIdx.x;               // 0..31
  const int qBase = blockIdx.y * 64;
  const float scale = 0.125f;                 // 64^-0.5

  const unsigned int* qu  = (const unsigned int*)qg  + (size_t)bh * SEQ * 32;
  const unsigned int* ku  = (const unsigned int*)kg  + (size_t)bh * SEQ * 32;
  const unsigned int* vTu = (const unsigned int*)vgT + (size_t)bh * SEQ * 32;

  // Q fragments (A-layout), K-dim = d: two 32-chunks
  ABFrag aQ[2];
  #pragma unroll
  for (int c = 0; c < 2; c++)
    #pragma unroll
    for (int j = 0; j < 8; j++)
      aQ[c].u[j] = qu[(size_t)(qBase + wave * 16 + ln) * 32 + c * 16 +
                      (j & 3) + hh * 4 + (j >> 2) * 8];

  float mrow[8], lrow[8];
  v8f oacc[4];
  #pragma unroll
  for (int j = 0; j < 8; j++) { mrow[j] = NEG_INF; lrow[j] = 0.0f; }
  #pragma unroll
  for (int t = 0; t < 4; t++)
    #pragma unroll
    for (int j = 0; j < 8; j++) oacc[t][j] = 0.0f;

  unsigned short* lds_p16 = (unsigned short*)lds_p;

  auto stage = [&](int kt, int buf) {
    #pragma unroll
    for (int i = 0; i < 4; i++) {
      int u = (i * 128 + tid) * 4;                       // 0..2044 step 4
      cp_b128(&lds_k[buf][u], &ku[(size_t)kt * 2048 + u]);
      int d = u >> 5, q = u & 31;                        // V: (d, keypair)
      cp_b128(&lds_v[buf][d * 32 + q], &vTu[(size_t)d * 1024 + kt * 32 + q]);
    }
  };

  stage(0, 0);
  cp_wait();
  __syncthreads();

  for (int kt = 0; kt < SEQ / 64; kt++) {
    const int cur = kt & 1;
    if (kt + 1 < SEQ / 64) stage(kt + 1, cur ^ 1);

    // ---- S = Q * K^T  (16 qrows x 64 keys per wave) ----
    v8f sacc[4];
    #pragma unroll
    for (int t = 0; t < 4; t++)
      #pragma unroll
      for (int j = 0; j < 8; j++) sacc[t][j] = 0.0f;
    #pragma unroll
    for (int t = 0; t < 4; t++) {
      #pragma unroll
      for (int c = 0; c < 2; c++) {
        ABFrag bf;
        #pragma unroll
        for (int j = 0; j < 8; j++)
          bf.u[j] = lds_k[cur][(t * 16 + ln) * 32 + c * 16 + (j & 3) + hh * 4 + (j >> 2) * 8];
        sacc[t] = __builtin_amdgcn_wmma_f32_16x16x32_bf16(
            false, aQ[c].v, false, bf.v, (short)0, sacc[t], false, false);
      }
    }

    // ---- online softmax; emit P (bf16) to LDS in (qrow, key) layout ----
    #pragma unroll
    for (int j = 0; j < 8; j++) {
      float rm = fmaxf(fmaxf(sacc[0][j], sacc[1][j]),
                       fmaxf(sacc[2][j], sacc[3][j]));
      rm = fmaxf(rm, __shfl_xor(rm, 1, 32));
      rm = fmaxf(rm, __shfl_xor(rm, 2, 32));
      rm = fmaxf(rm, __shfl_xor(rm, 4, 32));
      rm = fmaxf(rm, __shfl_xor(rm, 8, 32));
      float mnew = fmaxf(mrow[j], rm * scale);
      float fsc  = __expf(mrow[j] - mnew);
      mrow[j] = mnew;
      float p0 = __expf(sacc[0][j] * scale - mnew);
      float p1 = __expf(sacc[1][j] * scale - mnew);
      float p2 = __expf(sacc[2][j] * scale - mnew);
      float p3 = __expf(sacc[3][j] * scale - mnew);
      float rs = p0 + p1 + p2 + p3;
      rs += __shfl_xor(rs, 1, 32);
      rs += __shfl_xor(rs, 2, 32);
      rs += __shfl_xor(rs, 4, 32);
      rs += __shfl_xor(rs, 8, 32);
      lrow[j] = lrow[j] * fsc + rs;
      #pragma unroll
      for (int t = 0; t < 4; t++) oacc[t][j] *= fsc;
      int qr = wave * 16 + j + 8 * hh;
      lds_p16[qr * 64 + ln]      = f32_to_bf16(p0);
      lds_p16[qr * 64 + 16 + ln] = f32_to_bf16(p1);
      lds_p16[qr * 64 + 32 + ln] = f32_to_bf16(p2);
      lds_p16[qr * 64 + 48 + ln] = f32_to_bf16(p3);
    }
    __syncthreads();

    // ---- O += P * V ----
    #pragma unroll
    for (int c = 0; c < 2; c++) {
      ABFrag ap;
      #pragma unroll
      for (int j = 0; j < 8; j++)
        ap.u[j] = lds_p[(wave * 16 + ln) * 32 + c * 16 +
                        (j & 3) + hh * 4 + (j >> 2) * 8];
      #pragma unroll
      for (int t = 0; t < 4; t++) {
        ABFrag bv;
        #pragma unroll
        for (int j = 0; j < 8; j++)
          bv.u[j] = lds_v[cur][(t * 16 + ln) * 32 + c * 16 + (j & 3) + hh * 4 + (j >> 2) * 8];
        oacc[t] = __builtin_amdgcn_wmma_f32_16x16x32_bf16(
            false, ap.v, false, bv.v, (short)0, oacc[t], false, false);
      }
    }
    cp_wait();
    __syncthreads();
  }

  // ---- epilogue: O / l, write bf16 to (b, n, head*64 + d) ----
  const int b = bh >> 4, head = bh & 15;
  #pragma unroll
  for (int j = 0; j < 8; j++) {
    float inv = 1.0f / lrow[j];
    int n = qBase + wave * 16 + j + 8 * hh;
    size_t o = ((size_t)(b * SEQ + n)) * CDIM + head * HDIM;
    #pragma unroll
    for (int t = 0; t < 4; t++)
      outg[o + t * 16 + ln] = f32_to_bf16(oacc[t][j] * inv);
  }
}

// ---------------------------------------------------------------------------
// K3: projection GEMM (4096x1024 @ 1024x1024) + bias -> fp32 d_out
// ---------------------------------------------------------------------------
__global__ __launch_bounds__(128) void proj_gemm_kernel(
    const unsigned int* __restrict__ aa,    // attn out bf16 (4096 x 512) uints
    const unsigned int* __restrict__ wT,    // W_proj^T bf16 (1024 x 512) uints
    const float* __restrict__ bias,
    float* __restrict__ out) {
  __shared__ __align__(16) unsigned int lds_a[2][64 * 16];
  __shared__ __align__(16) unsigned int lds_b[2][64 * 16];

  const int tid  = threadIdx.x;
  const int wave = tid >> 5;
  const int lane = tid & 31;
  const int ln   = lane & 15;
  const int hh   = lane >> 4;
  const int rowBase = blockIdx.y * 64;
  const int colBase = blockIdx.x * 64;

  v8f acc[4];
  #pragma unroll
  for (int t = 0; t < 4; t++)
    #pragma unroll
    for (int j = 0; j < 8; j++) acc[t][j] = 0.0f;

  auto stage = [&](int kb, int buf) {
    const int kp0 = kb * 16;
    #pragma unroll
    for (int i = 0; i < 2; i++) {
      int u = i * 128 + tid;
      int r = u >> 2, q = (u & 3) * 4;
      cp_b128(&lds_a[buf][r * 16 + q],
              &aa[(size_t)(rowBase + r) * 512 + kp0 + q]);
      cp_b128(&lds_b[buf][r * 16 + q],
              &wT[(size_t)(colBase + r) * 512 + kp0 + q]);
    }
  };

  stage(0, 0);
  cp_wait();
  __syncthreads();

  for (int kb = 0; kb < 32; kb++) {
    const int cur = kb & 1;
    if (kb + 1 < 32) stage(kb + 1, cur ^ 1);

    ABFrag af;
    #pragma unroll
    for (int j = 0; j < 8; j++)
      af.u[j] = lds_a[cur][(wave * 16 + ln) * 16 + (j & 3) + hh * 4 + (j >> 2) * 8];
    #pragma unroll
    for (int t = 0; t < 4; t++) {
      ABFrag bf;
      #pragma unroll
      for (int j = 0; j < 8; j++)
        bf.u[j] = lds_b[cur][(t * 16 + ln) * 16 + (j & 3) + hh * 4 + (j >> 2) * 8];
      acc[t] = __builtin_amdgcn_wmma_f32_16x16x32_bf16(
          false, af.v, false, bf.v, (short)0, acc[t], false, false);
    }
    cp_wait();
    __syncthreads();
  }

  #pragma unroll
  for (int j = 0; j < 8; j++) {
    int m = rowBase + wave * 16 + j + 8 * hh;
    #pragma unroll
    for (int t = 0; t < 4; t++)
      out[(size_t)m * CDIM + colBase + t * 16 + ln] =
          acc[t][j] + bias[colBase + t * 16 + ln];
  }
}

// ---------------------------------------------------------------------------
// Launch
// ---------------------------------------------------------------------------
extern "C" void kernel_launch(void* const* d_in, const int* in_sizes, int n_in,
                              void* d_out, int out_size, void* d_ws, size_t ws_size,
                              hipStream_t stream) {
  const float* x        = (const float*)d_in[0];
  const float* rope_cos = (const float*)d_in[1];
  const float* rope_sin = (const float*)d_in[2];
  const float* W_qkv    = (const float*)d_in[3];
  const float* b_qkv    = (const float*)d_in[4];
  const float* W_proj   = (const float*)d_in[5];
  const float* b_proj   = (const float*)d_in[6];
  const float* q_norm_w = (const float*)d_in[7];
  const float* k_norm_w = (const float*)d_in[8];

  char* ws = (char*)d_ws;
  unsigned int*  xb     = (unsigned int*)(ws + 0);           //  8 MB
  unsigned int*  wqkvT  = (unsigned int*)(ws + 8388608);     //  6 MB (3072x512)
  unsigned int*  wprojT = (unsigned int*)(ws + 14680064);    //  2 MB (1024x512)
  unsigned short* qb    = (unsigned short*)(ws + 16777216);  //  8 MB
  unsigned short* kb    = (unsigned short*)(ws + 25165824);  //  8 MB
  unsigned short* vbT   = (unsigned short*)(ws + 33554432);  //  8 MB (bh,d,n)
  unsigned short* ab    = (unsigned short*)(ws + 41943040);  //  8 MB (ends 48 MB)

  int npx = MTOT * CDIM / 2;
  int npq = 3 * CDIM * (CDIM / 2);
  int npp = CDIM * (CDIM / 2);
  f32_to_bf16_kernel<<<(npx + 255) / 256, 256, 0, stream>>>(x, xb, npx);
  convert_transpose_kernel<<<(npq + 255) / 256, 256, 0, stream>>>(
      W_qkv, wqkvT, CDIM, 3 * CDIM);
  convert_transpose_kernel<<<(npp + 255) / 256, 256, 0, stream>>>(
      W_proj, wprojT, CDIM, CDIM);

  qkv_gemm_kernel<<<dim3(48, 64), 128, 0, stream>>>(
      xb, wqkvT, b_qkv, rope_cos, rope_sin, q_norm_w, k_norm_w, qb, kb, vbT);

  flash_attn_kernel<<<dim3(BATCH * NHEADS, SEQ / 64), 128, 0, stream>>>(
      qb, kb, vbT, ab);

  proj_gemm_kernel<<<dim3(16, 64), 128, 0, stream>>>(
      (const unsigned int*)ab, wprojT, b_proj, (float*)d_out);
}
